// PolicyNet_76905684402877
// MI455X (gfx1250) — matrix operand order repaired and checked by reading, not compiled, not run on hardware
//
#include <hip/hip_runtime.h>
#include <hip/hip_bf16.h>

// ---------------------------------------------------------------------------
// PolicyNet forward on gfx1250 (MI455X).  All GEMMs in bf16 via
// v_wmma_f32_16x16x32_bf16: 8 waves/block, 32x64 tile per wave (each LDS B
// fragment feeds two WMMAs), LDS-staged double-buffered 128-deep weight
// chunks; LN / softmax epilogues in fp32.
// ---------------------------------------------------------------------------

#define NBATCH 2048
#define MCB    8
#define KCW    256
#define DFEAT  256
#define LN_EPS 1e-6f

typedef __attribute__((ext_vector_type(16))) __bf16 v16bf;
typedef __attribute__((ext_vector_type(8)))  __bf16 v8bf;
typedef __attribute__((ext_vector_type(8)))  float  v8f;

// ---------------------------------------------------------------------------
// fp32 W[din,dout] -> bf16 W^T[dout,din]  (so GEMM B-fragments are contiguous
// along K).  Coalesced read, strided write; tiny cost vs GEMM.
// ---------------------------------------------------------------------------
__global__ void convert_wT(const float* __restrict__ w, __bf16* __restrict__ wt,
                           int din, int dout) {
  int idx = blockIdx.x * blockDim.x + threadIdx.x;
  int total = din * dout;
  if (idx >= total) return;
  int k = idx / dout;
  int n = idx - k * dout;
  wt[(size_t)n * din + k] = (__bf16)w[idx];
}

// ---------------------------------------------------------------------------
// fp32 (a [+ b]) -> bf16 into dst with row stride / column offset (used both
// for plain activation conversion, residual adds, and the r1|r2|r3|tr concat).
// ---------------------------------------------------------------------------
__global__ void prep_bf16(const float* __restrict__ a, const float* __restrict__ b,
                          __bf16* __restrict__ dst, int rows, int cols,
                          int dstStride, int colOff) {
  int idx = blockIdx.x * blockDim.x + threadIdx.x;
  int total = rows * cols;
  if (idx >= total) return;
  int r = idx / cols;
  int c = idx - r * cols;
  float v = a[idx];
  if (b) v += b[idx];
  dst[(size_t)r * dstStride + colOff + c] = (__bf16)v;
}

// ---------------------------------------------------------------------------
// C[N,Nout] = A[N,K](bf16) @ W(K,Nout)  with W given transposed as
// BT[Nout,K](bf16), bias fused.
//
// Block: 256 threads = 8 waves; block tile = 256 rows x 64 cols; each wave a
// 32x64 strip: 2 A fragments x 4 B fragments -> 8 accumulators.  Each LDS B
// fragment feeds TWO WMMAs (ds_loads per WMMA = 1, and every second WMMA can
// issue with no load dependency).  B is staged in LDS as 64x128 chunks
// (double-buffered, padded stride 136 bf16 -> bank-conflict-free, 16B-aligned
// b128 accesses) shared by all 8 waves; 32 WMMAs per barrier.  A fragments
// load straight from global (rows wave-private, L2-resident across colBase).
//
// Fragment layouts per CDNA5 ISA 7.12.2 (16-bit, 16x16x32):
//   A: lane m = lane%16, K = ksel..ksel+8 and ksel+16..ksel+24, ksel=8*(lane>=16)
//   B: lane n = lane%16, K = kb..kb+8 and kb+16..kb+24, kb=8*(lane>=16)
//   D: VGPR j, lane l -> row j + 8*(l>=16), col l%16
// ---------------------------------------------------------------------------
#define KCHUNK 128     // K-depth staged per barrier
#define SBS    136     // bf16 elems per staged B row (272B: 68 banks = 4 mod 64)

__global__ __launch_bounds__(256)
void gemm_bf16_wmma(const __bf16* __restrict__ A,   // [N, K]
                    const __bf16* __restrict__ BT,  // [Nout, K]
                    const float*  __restrict__ bias,
                    float* __restrict__ C,          // [N, Nout]
                    int N, int K, int Nout) {
  __shared__ __attribute__((aligned(32))) __bf16 sB[2][64 * SBS];  // 2 x 17KB

  const int tid  = threadIdx.x;
  const int lane = tid & 31;
  const int wave = tid >> 5;                   // 0..7
  const int m    = lane & 15;
  const int hi   = lane >> 4;                  // 0 or 1
  const int rowBase = blockIdx.y * 256 + wave * 32;
  const int colBase = blockIdx.x * 64;

  const __bf16* arow0 = A + (size_t)(rowBase +  0 + m) * K;
  const __bf16* arow1 = A + (size_t)(rowBase + 16 + m) * K;
  const int ksel = hi * 8;    // A-fragment K sub-select
  const int kb   = hi * 8;    // B-fragment K sub-select

  // Cooperative B loader: 64 tile-rows x 128 k-elems per chunk;
  // 4 threads/row, 32 contiguous elems (64B = 4 x b128) each.
  const int ldRow  = tid >> 2;        // 0..63
  const int ldBase = (tid & 3) * 32;  // 0,32,64,96
  const __bf16* bload = BT + (size_t)(colBase + ldRow) * K + ldBase;
  __bf16* sstore0 = &sB[0][ldRow * SBS + ldBase];
  __bf16* sstore1 = &sB[1][ldRow * SBS + ldBase];

  v8f acc[2][4];
#pragma unroll
  for (int g = 0; g < 2; ++g)
#pragma unroll
    for (int t = 0; t < 4; ++t) acc[g][t] = (v8f){};

  union Frag { v16bf v; v8bf h[2]; };

  // Compute one staged 128-deep chunk: 4 k-steps x (2 A-frags x 4 B-frags).
  auto compute = [&](const __bf16* sb, int kc) {
#pragma unroll
    for (int kk = 0; kk < KCHUNK; kk += 32) {
      Frag ua0, ua1;
      ua0.h[0] = *(const v8bf*)(arow0 + kc + kk + ksel);
      ua0.h[1] = *(const v8bf*)(arow0 + kc + kk + ksel + 16);
      ua1.h[0] = *(const v8bf*)(arow1 + kc + kk + ksel);
      ua1.h[1] = *(const v8bf*)(arow1 + kc + kk + ksel + 16);
#pragma unroll
      for (int t = 0; t < 4; ++t) {
        Frag ub;
        ub.h[0] = *(const v8bf*)(sb + (t * 16 + m) * SBS + kk + kb);
        ub.h[1] = *(const v8bf*)(sb + (t * 16 + m) * SBS + kk + kb + 16);
        acc[0][t] = __builtin_amdgcn_wmma_f32_16x16x32_bf16(
            false, ua0.v, false, ub.v, (short)0, acc[0][t], false, false);
        acc[1][t] = __builtin_amdgcn_wmma_f32_16x16x32_bf16(
            false, ua1.v, false, ub.v, (short)0, acc[1][t], false, false);
      }
    }
  };

  // Prime buffer 0 with chunk 0.
  {
    v8bf p0 = *(const v8bf*)(bload);
    v8bf p1 = *(const v8bf*)(bload + 8);
    v8bf p2 = *(const v8bf*)(bload + 16);
    v8bf p3 = *(const v8bf*)(bload + 24);
    *(v8bf*)(sstore0)      = p0;
    *(v8bf*)(sstore0 +  8) = p1;
    *(v8bf*)(sstore0 + 16) = p2;
    *(v8bf*)(sstore0 + 24) = p3;
  }
  __syncthreads();

  const int nChunks = K / KCHUNK;
  int buf = 0;
  for (int c = 0; c < nChunks - 1; ++c) {
    // Prefetch chunk c+1 into registers while computing chunk c.
    const __bf16* src = bload + (size_t)(c + 1) * KCHUNK;
    v8bf p0 = *(const v8bf*)(src);
    v8bf p1 = *(const v8bf*)(src + 8);
    v8bf p2 = *(const v8bf*)(src + 16);
    v8bf p3 = *(const v8bf*)(src + 24);

    compute(&sB[buf][0], c * KCHUNK);

    buf ^= 1;
    __bf16* dst = buf ? sstore1 : sstore0;
    *(v8bf*)(dst)      = p0;
    *(v8bf*)(dst +  8) = p1;
    *(v8bf*)(dst + 16) = p2;
    *(v8bf*)(dst + 24) = p3;
    __syncthreads();
  }
  compute(&sB[buf][0], (nChunks - 1) * KCHUNK);   // tail chunk, no prefetch

  // Epilogue: D row = rowBase + g*16 + j + 8*hi, col = colBase + 16t + m.
  float bias0 = bias[colBase +  0 + m];
  float bias1 = bias[colBase + 16 + m];
  float bias2 = bias[colBase + 32 + m];
  float bias3 = bias[colBase + 48 + m];
#pragma unroll
  for (int g = 0; g < 2; ++g) {
    const int rOff = rowBase + g * 16 + 8 * hi;
#pragma unroll
    for (int j = 0; j < 8; ++j) {
      float* crow = C + (size_t)(rOff + j) * Nout + colBase;
      crow[ 0 + m] = acc[g][0][j] + bias0;
      crow[16 + m] = acc[g][1][j] + bias1;
      crow[32 + m] = acc[g][2][j] + bias2;
      crow[48 + m] = acc[g][3][j] + bias3;
    }
  }
}

// ---------------------------------------------------------------------------
// out = LayerNorm(relu(in)) * g + be, one block (256 thr) per row.
// ---------------------------------------------------------------------------
__global__ __launch_bounds__(256)
void relu_ln(const float* __restrict__ in, const float* __restrict__ g,
             const float* __restrict__ be, float* __restrict__ out, int cols) {
  const int row = blockIdx.x;
  const float* r = in + (size_t)row * cols;
  float s = 0.f, s2 = 0.f;
  for (int c = threadIdx.x; c < cols; c += 256) {
    float h = fmaxf(r[c], 0.f);
    s += h; s2 += h * h;
  }
  __shared__ float sh1[256], sh2[256];
  sh1[threadIdx.x] = s; sh2[threadIdx.x] = s2;
  __syncthreads();
  for (int o = 128; o > 0; o >>= 1) {
    if (threadIdx.x < o) {
      sh1[threadIdx.x] += sh1[threadIdx.x + o];
      sh2[threadIdx.x] += sh2[threadIdx.x + o];
    }
    __syncthreads();
  }
  const float inv_n = 1.f / (float)cols;
  const float mu  = sh1[0] * inv_n;
  const float var = sh2[0] * inv_n - mu * mu;
  const float inv = rsqrtf(var + LN_EPS);
  float* o = out + (size_t)row * cols;
  for (int c = threadIdx.x; c < cols; c += 256) {
    float h = fmaxf(r[c], 0.f);
    o[c] = (h - mu) * inv * g[c] + be[c];
  }
}

// ---------------------------------------------------------------------------
// Per-row log-softmax over K=256 logits; accumulate entropy and -log p(b).
// One block (256 thr == K) per row; single writer per row so plain RMW.
// ---------------------------------------------------------------------------
__global__ __launch_bounds__(256)
void logits_reduce(const float* __restrict__ logits, const int* __restrict__ bsel,
                   int cb, float* __restrict__ nlpOut, float* __restrict__ entOut) {
  const int row = blockIdx.x;
  const int t = threadIdx.x;
  const float x = logits[(size_t)row * KCW + t];
  __shared__ float sh[256];

  sh[t] = x; __syncthreads();
  for (int o = 128; o > 0; o >>= 1) {
    if (t < o) sh[t] = fmaxf(sh[t], sh[t + o]);
    __syncthreads();
  }
  const float mx = sh[0];
  __syncthreads();

  const float e = __expf(x - mx);
  sh[t] = e; __syncthreads();
  for (int o = 128; o > 0; o >>= 1) {
    if (t < o) sh[t] += sh[t + o];
    __syncthreads();
  }
  const float sum = sh[0];
  __syncthreads();

  const float ls = x - mx - __logf(sum);
  sh[t] = (e / sum) * ls; __syncthreads();
  for (int o = 128; o > 0; o >>= 1) {
    if (t < o) sh[t] += sh[t + o];
    __syncthreads();
  }
  if (t == 0) entOut[row] += -sh[0];

  const int bi = bsel[row * MCB + cb];
  if (t == bi) nlpOut[row] += -ls;
}

// ---------------------------------------------------------------------------
// Initialize d_out: [N*M] b-as-float, [N] nlp=0, [N] ent=0.
// ---------------------------------------------------------------------------
__global__ void init_out(const int* __restrict__ b, float* __restrict__ out) {
  int idx = blockIdx.x * blockDim.x + threadIdx.x;
  const int nm = NBATCH * MCB;
  const int total = nm + 2 * NBATCH;
  if (idx >= total) return;
  out[idx] = (idx < nm) ? (float)b[idx] : 0.f;
}

// ---------------------------------------------------------------------------
// Host orchestration
// ---------------------------------------------------------------------------
static inline int ceil_div(int a, int b) { return (a + b - 1) / b; }

extern "C" void kernel_launch(void* const* d_in, const int* in_sizes, int n_in,
                              void* d_out, int out_size, void* d_ws, size_t ws_size,
                              hipStream_t stream) {
  (void)in_sizes; (void)n_in; (void)out_size; (void)ws_size;
  const float* x    = (const float*)d_in[0];     // [N, D]
  const int*   bsel = (const int*)d_in[1];       // [N, M]
  // d_in[2] = greedy (ignored; b is provided)
  float* out = (float*)d_out;
  float* nlpOut = out + NBATCH * MCB;
  float* entOut = nlpOut + NBATCH;

  // Workspace carve-up -------------------------------------------------------
  char* p = (char*)d_ws;
  auto takeF = [&](size_t n) { float* q = (float*)p; p += n * sizeof(float); return q; };
  auto takeB = [&](size_t n) { __bf16* q = (__bf16*)p; p += n * sizeof(__bf16); return q; };
  float*  gemmOut = takeF((size_t)NBATCH * 2048);  // generic GEMM output
  float*  preO = takeF((size_t)NBATCH * 256);
  float*  l1O  = takeF((size_t)NBATCH * 1024);
  float*  l2O  = takeF((size_t)NBATCH * 512);
  float*  l3O  = takeF((size_t)NBATCH * 256);
  float*  trO  = takeF((size_t)NBATCH * 256);
  float*  r3O  = takeF((size_t)NBATCH * 512);
  float*  r2O  = takeF((size_t)NBATCH * 1024);
  float*  r1O  = takeF((size_t)NBATCH * 256);
  float*  lgO  = takeF((size_t)NBATCH * 2048);
  __bf16* act  = takeB((size_t)NBATCH * 2048);     // bf16 GEMM input
  __bf16* wt   = takeB((size_t)2048 * 2048);       // bf16 W^T staging

  // Helpers ------------------------------------------------------------------
  auto prep = [&](const float* a, const float* bres, int cols, int stride, int off) {
    int total = NBATCH * cols;
    prep_bf16<<<ceil_div(total, 256), 256, 0, stream>>>(a, bres, act, NBATCH, cols,
                                                        stride, off);
  };
  auto gemm = [&](const float* wfp, const float* bias, float* dst, int din, int dout) {
    convert_wT<<<ceil_div(din * dout, 256), 256, 0, stream>>>(wfp, wt, din, dout);
    dim3 grid(dout / 64, NBATCH / 256);
    gemm_bf16_wmma<<<grid, 256, 0, stream>>>(act, wt, bias, dst, NBATCH, din, dout);
  };
  auto ln = [&](const float* g, const float* be, float* dst, int cols) {
    relu_ln<<<NBATCH, 256, 0, stream>>>(gemmOut, g, be, dst, cols);
  };

  init_out<<<ceil_div(NBATCH * MCB + 2 * NBATCH, 256), 256, 0, stream>>>(bsel, out);

  // Param leaf layout per codebook (jax sorted-dict flatten order), 40 leaves:
  //  0:head.b 1:head.w 2:lg.be 3:lg.g 4:lg.lin.b 5:lg.lin.w
  //  6:fin.b 7:fin.w
  //  8..11:l1{be,g,b,w} 12..15:l2 16..19:l3 20..23:pre
  // 24..27:r1 28..31:r2 32..35:r3 36..39:tr
  for (int cb = 0; cb < MCB; ++cb) {
    const int base = 3 + 40 * cb;
    auto P = [&](int off) { return (const float*)d_in[base + off]; };
    const float *head_b = P(0),  *head_w = P(1);
    const float *lg_be  = P(2),  *lg_g   = P(3),  *lg_b = P(4),  *lg_w = P(5);
    const float *fin_b  = P(6),  *fin_w  = P(7);
    const float *l1_be = P(8),  *l1_g = P(9),  *l1_b = P(10), *l1_w = P(11);
    const float *l2_be = P(12), *l2_g = P(13), *l2_b = P(14), *l2_w = P(15);
    const float *l3_be = P(16), *l3_g = P(17), *l3_b = P(18), *l3_w = P(19);
    const float *pr_be = P(20), *pr_g = P(21), *pr_b = P(22), *pr_w = P(23);
    const float *r1_be = P(24), *r1_g = P(25), *r1_b = P(26), *r1_w = P(27);
    const float *r2_be = P(28), *r2_g = P(29), *r2_b = P(30), *r2_w = P(31);
    const float *r3_be = P(32), *r3_g = P(33), *r3_b = P(34), *r3_w = P(35);
    const float *tr_be = P(36), *tr_g = P(37), *tr_b = P(38), *tr_w = P(39);

    // pyramid
    prep(x, nullptr, DFEAT, DFEAT, 0);
    gemm(pr_w, pr_b, gemmOut, 256, 256);   ln(pr_g, pr_be, preO, 256);
    prep(preO, nullptr, 256, 256, 0);
    gemm(l1_w, l1_b, gemmOut, 256, 1024);  ln(l1_g, l1_be, l1O, 1024);
    prep(l1O, nullptr, 1024, 1024, 0);
    gemm(l2_w, l2_b, gemmOut, 1024, 512);  ln(l2_g, l2_be, l2O, 512);
    prep(l2O, nullptr, 512, 512, 0);
    gemm(l3_w, l3_b, gemmOut, 512, 256);   ln(l3_g, l3_be, l3O, 256);
    prep(l3O, nullptr, 256, 256, 0);
    gemm(tr_w, tr_b, gemmOut, 256, 256);   ln(tr_g, tr_be, trO, 256);
    prep(trO, preO, 256, 256, 0);                       // tr + x(pre)
    gemm(r3_w, r3_b, gemmOut, 256, 512);   ln(r3_g, r3_be, r3O, 512);
    prep(r3O, l2O, 512, 512, 0);                        // r3 + l2
    gemm(r2_w, r2_b, gemmOut, 512, 1024);  ln(r2_g, r2_be, r2O, 1024);
    prep(r2O, l1O, 1024, 1024, 0);                      // r2 + l1
    gemm(r1_w, r1_b, gemmOut, 1024, 256);  ln(r1_g, r1_be, r1O, 256);

    // fin on concat([r1, r2, r3, tr]) -> [N, 2048]
    prep(r1O, nullptr,  256, 2048,    0);
    prep(r2O, nullptr, 1024, 2048,  256);
    prep(r3O, nullptr,  512, 2048, 1280);
    prep(trO, nullptr,  256, 2048, 1792);
    gemm(fin_w, fin_b, gemmOut, 2048, 2048);            // plain Linear

    // lg (LayerGroup 2048 -> 2048)
    prep(gemmOut, nullptr, 2048, 2048, 0);
    gemm(lg_w, lg_b, gemmOut, 2048, 2048); ln(lg_g, lg_be, lgO, 2048);

    // head (Linear 2048 -> 256) + softmax statistics
    prep(lgO, nullptr, 2048, 2048, 0);
    gemm(head_w, head_b, gemmOut, 2048, 256);
    logits_reduce<<<NBATCH, 256, 0, stream>>>(gemmOut, bsel, cb, nlpOut, entOut);
  }
}
